// DCRNNBaseline_88201448391515
// MI455X (gfx1250) — matrix operand order repaired and checked by reading, not compiled, not run on hardware
//
#include <hip/hip_runtime.h>
#include <hip/hip_bf16.h>

// ---------------------------------------------------------------------------
// DCRNN forward for MI455X (gfx1250), bf16 WMMA everywhere.
//
// B=128, T=12, N=2000, IN=2, HID=64, PRED=12.
// Dominant cost: adj(2000x2000) x V(2000x8448) per phase (2 phases x 12 steps)
// ~1.62 TFLOP + dense 132->{128,64} layers ~0.19 TFLOP. Working set is L2
// resident (192MB) -> matrix-core bound -> v_wmma_f32_16x16x32_bf16.
//
// Workspace layout (bytes, 256-aligned offsets), total ~303.2 MB:
//   adjb   bf16 [2000][2048]      @ 0
//   Acat1  bf16 [256000][160]     @ 8,192,000
//   Acat2  bf16 [256000][160]     @ 90,112,000
//   h      f32  [256000][64]      @ 172,032,000
//   z      f32  [256000][64]      @ 237,568,000
//   Wg     bf16 [160][128]        @ 303,104,000
//   Wc     bf16 [160][64]         @ 303,144,960
// rIdx = n*128 + b. Acat cols: [0..1]=x_t, [2..65]=h (or r*h),
// [66..131]=adj-mix output, [132..159]=zero K-padding.
// ---------------------------------------------------------------------------

#define BATCH 128
#define TSTEPS 12
#define NNODE 2000
#define HID 64
#define PRED 12
#define KPAD 2048          // adj K padded to multiple of 32
#define CDIM 8448          // BATCH * 66
#define AST 160            // Acat row stride (elements)
#define ROWS 256000        // NNODE * BATCH
#define KITERS (KPAD / 32) // 64

typedef __attribute__((ext_vector_type(16))) __bf16 v16bf;
typedef __attribute__((ext_vector_type(8)))  __bf16 v8bf;
typedef __attribute__((ext_vector_type(8)))  float  v8f;
typedef __attribute__((ext_vector_type(8)))  unsigned short us8;
typedef __attribute__((ext_vector_type(4)))  float  f4;

union V16U { v16bf v; v8bf h[2]; };

static __device__ __forceinline__ unsigned short f2bf(float f) {
    unsigned int u = __float_as_uint(f);
    u += 0x7FFFu + ((u >> 16) & 1u);   // round-to-nearest-even
    return (unsigned short)(u >> 16);
}

static __device__ __forceinline__ v8f wmma_bf16(v16bf a, v16bf b, v8f c) {
    return __builtin_amdgcn_wmma_f32_16x16x32_bf16(
        /*neg_a=*/false, a, /*neg_b=*/false, b,
        /*c_mod=*/(short)0, c, /*reuse_a=*/false, /*reuse_b=*/false);
}

// ---------------------------------------------------------------- utilities
__global__ void k_zero(unsigned int* p, size_t n32) {
    size_t i = (size_t)blockIdx.x * 256 + threadIdx.x;
    size_t stride = (size_t)gridDim.x * 256;
    for (; i < n32; i += stride) p[i] = 0u;
}

__global__ void k_adj_convert(const float* __restrict__ adj,
                              unsigned short* __restrict__ adjb) {
    int idx = blockIdx.x * 256 + threadIdx.x;          // 0 .. 2000*2048-1
    int n = idx >> 11;
    int m = idx & (KPAD - 1);
    float v = (m < NNODE) ? adj[(size_t)n * NNODE + m] : 0.0f;
    adjb[idx] = f2bf(v);
}

__global__ void k_w_convert(const float* __restrict__ Wg,
                            const float* __restrict__ Wc,
                            unsigned short* __restrict__ Wgb,
                            unsigned short* __restrict__ Wcb) {
    int idx = blockIdx.x * 256 + threadIdx.x;          // 0 .. 30719
    if (idx < 160 * 128) {
        int k = idx >> 7, c = idx & 127;
        Wgb[idx] = (k < 132) ? f2bf(Wg[k * 128 + c]) : (unsigned short)0;
    } else {
        int j = idx - 160 * 128;
        if (j < 160 * 64) {
            int k = j >> 6, c = j & 63;
            Wcb[j] = (k < 132) ? f2bf(Wc[k * 64 + c]) : (unsigned short)0;
        }
    }
}

// Write x_t into both Acat buffers and h into Acat1 (bf16, u32-paired stores).
__global__ void k_pack(const float* __restrict__ x,
                       const float* __restrict__ h,
                       unsigned short* __restrict__ Acat1,
                       unsigned short* __restrict__ Acat2, int t) {
    int r = blockIdx.x * 256 + threadIdx.x;            // rIdx
    int n = r >> 7, b = r & 127;
    const float* xp = x + (((size_t)b * TSTEPS + t) * NNODE + n) * 2;
    unsigned int xpair =
        (unsigned int)f2bf(xp[0]) | ((unsigned int)f2bf(xp[1]) << 16);
    unsigned int* a1 = (unsigned int*)(Acat1 + (size_t)r * AST);
    unsigned int* a2 = (unsigned int*)(Acat2 + (size_t)r * AST);
    a1[0] = xpair;
    a2[0] = xpair;
    const float* hp = h + (size_t)r * HID;
    #pragma unroll
    for (int j = 0; j < 32; ++j) {
        unsigned int pr = (unsigned int)f2bf(hp[2 * j]) |
                          ((unsigned int)f2bf(hp[2 * j + 1]) << 16);
        a1[1 + j] = pr;
    }
}

// ---------------------------------------------------- adj-mix GEMM (WMMA)
// C[n, c] = sum_m adjb[n, m] * V[m, c], V read from Acat cols 0..65 via
// (m, c=bcol*66+d) -> Acat[(m*128+bcol)*160 + d]; result stored bf16 into
// Acat cols 66..131. Block tile 64x128, 8 waves, register-staged global
// loads + double-buffered LDS, one barrier per K-iteration.
__global__ __launch_bounds__(256) void k_adjmix(
        const unsigned short* __restrict__ adjb,
        unsigned short* __restrict__ Acat) {
    __shared__ unsigned short ldsA[2][64 * 40];   // [buf][row][k]
    __shared__ unsigned short ldsB[2][128 * 40];  // [buf][c_local][k]

    const int tid  = threadIdx.x;
    const int wave = tid >> 5;
    const int lane = tid & 31;
    const int rowbase = blockIdx.x * 64;       // node rows (M)
    const int colbase = blockIdx.y * 128;      // c columns (N)
    const int wr = wave >> 1;                  // row block 0..3
    const int wc = wave & 1;                   // col half 0..1

    // --- staging metadata (branchless guards via clamp + select)
    const int arow_l = tid >> 2;
    const int akcol  = (tid & 3) * 8;
    const int gr     = rowbase + arow_l;
    const bool aok   = gr < NNODE;
    const unsigned short* aptr =
        adjb + (size_t)(aok ? gr : 0) * KPAD + akcol;

    int boffG[8], ldsBoff[8], kbase[8];
    #pragma unroll
    for (int j = 0; j < 8; ++j) {
        int idx  = tid + j * 256;              // 0..2047
        int cl   = idx & 127;
        int kp   = idx >> 7;                   // 0..15
        int c    = colbase + cl;
        int bcol = c / 66;
        int d    = c - bcol * 66;
        boffG[j]   = bcol * AST + d;
        ldsBoff[j] = cl * 40 + kp * 2;
        kbase[j]   = kp * 2;
    }

    us8 aR;
    unsigned int bR[8];
    auto stage = [&](int kk) {
        aR = *(const us8*)(aptr + kk);
        #pragma unroll
        for (int j = 0; j < 8; ++j) {
            int k   = kk + kbase[j];
            bool ok = k < NNODE;               // k even => k,k+1 both ok
            int kc  = ok ? k : 0;
            const unsigned short* s =
                Acat + (size_t)kc * (128 * AST) + boffG[j];
            unsigned int lo = s[0];
            unsigned int hi = s[128 * AST];
            unsigned int pr = lo | (hi << 16);
            bR[j] = ok ? pr : 0u;
        }
    };

    v8f acc[4] = {};
    stage(0);

    const int rl   = wr * 16 + (lane & 15);
    const int ko8  = (lane >> 4) * 8;
    const int ko16 = (lane >> 4) * 16;
    const int cl0  = wc * 64 + (lane & 15);

    for (int it = 0; it < KITERS; ++it) {
        const int buf = it & 1;
        // commit staged registers to LDS
        {
            us8 zv = {};
            *(us8*)(&ldsA[buf][arow_l * 40 + akcol]) = aok ? aR : zv;
            #pragma unroll
            for (int j = 0; j < 8; ++j)
                *(unsigned int*)(&ldsB[buf][ldsBoff[j]]) = bR[j];
        }
        // issue next panel's global loads (latency hidden under compute)
        if (it + 1 < KITERS) stage((it + 1) * 32);
        __syncthreads();
        // batched fragment loads, then WMMA chain
        V16U a, b0, b1, b2, b3;
        a.h[0]  = *(const v8bf*)(&ldsA[buf][rl * 40 + ko16]);
        a.h[1]  = *(const v8bf*)(&ldsA[buf][rl * 40 + ko16 + 8]);
        b0.h[0] = *(const v8bf*)(&ldsB[buf][(cl0     ) * 40 + ko16]);
        b0.h[1] = *(const v8bf*)(&ldsB[buf][(cl0     ) * 40 + ko16 + 8]);
        b1.h[0] = *(const v8bf*)(&ldsB[buf][(cl0 + 16) * 40 + ko16]);
        b1.h[1] = *(const v8bf*)(&ldsB[buf][(cl0 + 16) * 40 + ko16 + 8]);
        b2.h[0] = *(const v8bf*)(&ldsB[buf][(cl0 + 32) * 40 + ko16]);
        b2.h[1] = *(const v8bf*)(&ldsB[buf][(cl0 + 32) * 40 + ko16 + 8]);
        b3.h[0] = *(const v8bf*)(&ldsB[buf][(cl0 + 48) * 40 + ko16]);
        b3.h[1] = *(const v8bf*)(&ldsB[buf][(cl0 + 48) * 40 + ko16 + 8]);
        acc[0] = wmma_bf16(a.v, b0.v, acc[0]);
        acc[1] = wmma_bf16(a.v, b1.v, acc[1]);
        acc[2] = wmma_bf16(a.v, b2.v, acc[2]);
        acc[3] = wmma_bf16(a.v, b3.v, acc[3]);
    }
    // --- epilogue: scatter bf16 into Acat cols 66..131
    #pragma unroll
    for (int ct = 0; ct < 4; ++ct) {
        #pragma unroll
        for (int i = 0; i < 8; ++i) {
            int rlo = wr * 16 + (lane >> 4) * 8 + i;
            int n   = rowbase + rlo;
            if (n < NNODE) {
                int c    = colbase + wc * 64 + ct * 16 + (lane & 15);
                int bcol = c / 66;
                int d    = c - bcol * 66;
                Acat[((size_t)(n * 128 + bcol)) * AST + 66 + d] =
                    f2bf(acc[ct][i]);
            }
        }
    }
}

// ---------------------------------------------------- gate GEMM + sigmoid
__global__ __launch_bounds__(256) void k_gate(
        const unsigned short* __restrict__ Acat1,
        const unsigned short* __restrict__ Wgb,
        const float* __restrict__ bg,
        const float* __restrict__ h,
        float* __restrict__ z,
        unsigned short* __restrict__ Acat2) {
    __shared__ unsigned short ldsW[128 * 160];     // [col][k]
    const int tid = threadIdx.x, wave = tid >> 5, lane = tid & 31;
    const size_t rowbase = (size_t)blockIdx.x * 128;

    #pragma unroll 4
    for (int i = 0; i < 80; ++i) {
        int idx = tid + i * 256;                   // 0..20479
        int k = idx >> 7, c = idx & 127;
        ldsW[c * 160 + k] = Wgb[idx];
    }

    // hoist all A-fragment global loads (memory-level parallelism)
    const size_t rIdx = rowbase + wave * 16 + (lane & 15);
    const unsigned short* arow = Acat1 + rIdx * AST;
    V16U afr[5];
    #pragma unroll
    for (int kc5 = 0; kc5 < 5; ++kc5) {
        int koff = kc5 * 32 + (lane >> 4) * 16;
        afr[kc5].h[0] = *(const v8bf*)(arow + koff);
        afr[kc5].h[1] = *(const v8bf*)(arow + koff + 8);
    }
    __syncthreads();

    v8f acc[8] = {};
    #pragma unroll
    for (int kc5 = 0; kc5 < 5; ++kc5) {
        int ko = kc5 * 32 + (lane >> 4) * 16;
        #pragma unroll
        for (int ct = 0; ct < 8; ++ct) {
            V16U b;
            int cl = ct * 16 + (lane & 15);
            b.h[0] = *(const v8bf*)(&ldsW[cl * 160 + ko]);
            b.h[1] = *(const v8bf*)(&ldsW[cl * 160 + ko + 8]);
            acc[ct] = wmma_bf16(afr[kc5].v, b.v, acc[ct]);
        }
    }
    #pragma unroll
    for (int ct = 0; ct < 8; ++ct) {
        #pragma unroll
        for (int i = 0; i < 8; ++i) {
            size_t r = rowbase + wave * 16 + (lane >> 4) * 8 + i;
            int col  = ct * 16 + (lane & 15);
            float pre = acc[ct][i] + bg[col];
            float s   = 1.0f / (1.0f + __expf(-pre));
            if (col < 64) {
                z[r * 64 + col] = s;
            } else {
                int j = col - 64;
                float rh = s * h[r * 64 + j];
                Acat2[r * AST + 2 + j] = f2bf(rh);
            }
        }
    }
}

// ---------------------------------------------------- cand GEMM + GRU update
__global__ __launch_bounds__(256) void k_cand(
        const unsigned short* __restrict__ Acat2,
        const unsigned short* __restrict__ Wcb,
        const float* __restrict__ bc,
        const float* __restrict__ z,
        float* __restrict__ h) {
    __shared__ unsigned short ldsW[64 * 160];      // [col][k]
    const int tid = threadIdx.x, wave = tid >> 5, lane = tid & 31;
    const size_t rowbase = (size_t)blockIdx.x * 128;

    #pragma unroll 4
    for (int i = 0; i < 40; ++i) {
        int idx = tid + i * 256;                   // 0..10239
        int k = idx >> 6, c = idx & 63;
        ldsW[c * 160 + k] = Wcb[idx];
    }

    const size_t rIdx = rowbase + wave * 16 + (lane & 15);
    const unsigned short* arow = Acat2 + rIdx * AST;
    V16U afr[5];
    #pragma unroll
    for (int kc5 = 0; kc5 < 5; ++kc5) {
        int koff = kc5 * 32 + (lane >> 4) * 16;
        afr[kc5].h[0] = *(const v8bf*)(arow + koff);
        afr[kc5].h[1] = *(const v8bf*)(arow + koff + 8);
    }
    __syncthreads();

    v8f acc[4] = {};
    #pragma unroll
    for (int kc5 = 0; kc5 < 5; ++kc5) {
        int ko = kc5 * 32 + (lane >> 4) * 16;
        #pragma unroll
        for (int ct = 0; ct < 4; ++ct) {
            V16U b;
            int cl = ct * 16 + (lane & 15);
            b.h[0] = *(const v8bf*)(&ldsW[cl * 160 + ko]);
            b.h[1] = *(const v8bf*)(&ldsW[cl * 160 + ko + 8]);
            acc[ct] = wmma_bf16(afr[kc5].v, b.v, acc[ct]);
        }
    }
    #pragma unroll
    for (int ct = 0; ct < 4; ++ct) {
        #pragma unroll
        for (int i = 0; i < 8; ++i) {
            size_t r = rowbase + wave * 16 + (lane >> 4) * 8 + i;
            int col  = ct * 16 + (lane & 15);
            float ht = tanhf(acc[ct][i] + bc[col]);
            float zv = z[r * 64 + col];
            float hv = h[r * 64 + col];
            h[r * 64 + col] = (1.0f - zv) * hv + zv * ht;
        }
    }
}

// ---------------------------------------------------- readout (coalesced out)
__global__ void k_readout(const float* __restrict__ h,
                          const float* __restrict__ Wro,
                          const float* __restrict__ bro,
                          float* __restrict__ out) {
    int idx = blockIdx.x * 256 + threadIdx.x;       // 0 .. ROWS*PRED-1
    int q = idx / NNODE;                            // b*PRED + p
    int n = idx - q * NNODE;
    int b = q / PRED;
    int p = q - b * PRED;
    const float* hp = h + ((size_t)(n * 128 + b)) * HID;
    float s = bro[p];
    #pragma unroll
    for (int j4 = 0; j4 < HID / 4; ++j4) {
        f4 v = *(const f4*)(hp + j4 * 4);
        s += v.x * Wro[(j4 * 4 + 0) * PRED + p];
        s += v.y * Wro[(j4 * 4 + 1) * PRED + p];
        s += v.z * Wro[(j4 * 4 + 2) * PRED + p];
        s += v.w * Wro[(j4 * 4 + 3) * PRED + p];
    }
    out[(size_t)q * NNODE + n] = s;                 // coalesced
}

// ---------------------------------------------------------------------------
extern "C" void kernel_launch(void* const* d_in, const int* in_sizes, int n_in,
                              void* d_out, int out_size, void* d_ws, size_t ws_size,
                              hipStream_t stream) {
    (void)in_sizes; (void)n_in; (void)out_size; (void)ws_size;
    const float* x      = (const float*)d_in[0];
    const float* adj    = (const float*)d_in[1];
    const float* W_gate = (const float*)d_in[2];
    const float* b_gate = (const float*)d_in[3];
    const float* W_cand = (const float*)d_in[4];
    const float* b_cand = (const float*)d_in[5];
    const float* W_ro   = (const float*)d_in[6];
    const float* b_ro   = (const float*)d_in[7];
    float* out = (float*)d_out;

    char* ws = (char*)d_ws;
    unsigned short* adjb  = (unsigned short*)(ws + 0);
    unsigned short* Acat1 = (unsigned short*)(ws + 8192000);
    unsigned short* Acat2 = (unsigned short*)(ws + 90112000);
    float*          hbuf  = (float*)(ws + 172032000);
    float*          zbuf  = (float*)(ws + 237568000);
    unsigned short* Wgb   = (unsigned short*)(ws + 303104000);
    unsigned short* Wcb   = (unsigned short*)(ws + 303144960);

    // zero h and both Acat buffers (K-padding cols must stay zero)
    k_zero<<<2048, 256, 0, stream>>>((unsigned int*)hbuf,
                                     (size_t)ROWS * HID);
    k_zero<<<2048, 256, 0, stream>>>((unsigned int*)Acat1,
                                     (size_t)ROWS * AST / 2);
    k_zero<<<2048, 256, 0, stream>>>((unsigned int*)Acat2,
                                     (size_t)ROWS * AST / 2);

    k_adj_convert<<<(NNODE * KPAD) / 256, 256, 0, stream>>>(adj, adjb);
    k_w_convert<<<120, 256, 0, stream>>>(W_gate, W_cand, Wgb, Wcb);

    for (int t = 0; t < TSTEPS; ++t) {
        k_pack<<<ROWS / 256, 256, 0, stream>>>(x, hbuf, Acat1, Acat2, t);
        k_adjmix<<<dim3(32, CDIM / 128), 256, 0, stream>>>(adjb, Acat1);
        k_gate<<<ROWS / 128, 256, 0, stream>>>(Acat1, Wgb, b_gate, hbuf,
                                               zbuf, Acat2);
        k_adjmix<<<dim3(32, CDIM / 128), 256, 0, stream>>>(adjb, Acat2);
        k_cand<<<ROWS / 128, 256, 0, stream>>>(Acat2, Wcb, b_cand, zbuf, hbuf);
    }
    k_readout<<<(ROWS * PRED) / 256, 256, 0, stream>>>(hbuf, W_ro, b_ro, out);
}
